// GraphTransformerLayer_56667798503547
// MI455X (gfx1250) — matrix operand order repaired and checked by reading, not compiled, not run on hardware
//
#include <hip/hip_runtime.h>
#include <hip/hip_bf16.h>
#include <math.h>

typedef _Float16 v16h __attribute__((ext_vector_type(16)));
typedef _Float16 v8h  __attribute__((ext_vector_type(8)));
typedef float    v8f  __attribute__((ext_vector_type(8)));

#define D_MODEL 512
#define NHEADS  8
#define DK      64
#define NTOK    1024
#define BATCH   4
#define MROWS   4096          // B * N
#define DFF     2048

// ---------------------------------------------------------------------------
// WMMA fragment helpers (CDNA5 16x16x32 f16 -> f32)
// A 16x32 layout: lane L -> row L%16; K chunks {0..7,16..23} (L<16) or
//                 {8..15,24..31} (L>=16): two contiguous 16B loads.
// B 32x16 layout: lane L -> col L%16; same K chunks as A ({0..7,16..23} or
//                 {8..15,24..31}): two 16B loads from a [N,K] row-major matrix (B^T).
// C/D: reg r -> row r + 8*(L>=16), col L%16.
// ---------------------------------------------------------------------------
__device__ __forceinline__ v16h load_a16(const _Float16* A, int lda, int m0,
                                         int k0, int lane) {
  const _Float16* p = A + (size_t)(m0 + (lane & 15)) * lda + k0 + ((lane & 16) ? 8 : 0);
  v16h a;
  ((v8h*)&a)[0] = *(const v8h*)(p);
  ((v8h*)&a)[1] = *(const v8h*)(p + 16);
  return a;
}

__device__ __forceinline__ v16h load_b16(const _Float16* W, int ldw, int n0,
                                         int k0, int lane) {
  const _Float16* p = W + (size_t)(n0 + (lane & 15)) * ldw + k0 + ((lane & 16) ? 8 : 0);
  v16h b;
  ((v8h*)&b)[0] = *(const v8h*)(p);
  ((v8h*)&b)[1] = *(const v8h*)(p + 16);
  return b;
}

__device__ __forceinline__ v8f wmma_f16(v16h a, v16h b, v8f c) {
  return __builtin_amdgcn_wmma_f32_16x16x32_f16(false, a, false, b, (short)0, c,
                                                false, false);
}

// ---------------------------------------------------------------------------
// fp32 -> fp16 convert
// ---------------------------------------------------------------------------
__global__ void f32_to_f16_kernel(const float* __restrict__ in,
                                  _Float16* __restrict__ out, int n) {
  int i = blockIdx.x * blockDim.x + threadIdx.x;
  if (i < n) out[i] = (_Float16)in[i];
}

// ---------------------------------------------------------------------------
// LayerNorm: one wave per 512-wide row, f32 in -> f16 out
// ---------------------------------------------------------------------------
__global__ __launch_bounds__(256) void ln_kernel(const float* __restrict__ x,
                                                 const float* __restrict__ g,
                                                 const float* __restrict__ bn,
                                                 _Float16* __restrict__ out,
                                                 int rows) {
  int lane = threadIdx.x & 31;
  int row = blockIdx.x * (blockDim.x >> 5) + (threadIdx.x >> 5);
  if (row >= rows) return;
  const float* xr = x + (size_t)row * D_MODEL;
  float v[16], s = 0.f, ss = 0.f;
#pragma unroll
  for (int i = 0; i < 16; ++i) {
    v[i] = xr[lane + 32 * i];
    s += v[i];
    ss += v[i] * v[i];
  }
#pragma unroll
  for (int m = 1; m < 32; m <<= 1) {
    s += __shfl_xor(s, m, 32);
    ss += __shfl_xor(ss, m, 32);
  }
  float mean = s * (1.f / 512.f);
  float var = ss * (1.f / 512.f) - mean * mean;
  float rstd = rsqrtf(var + 1e-5f);
  _Float16* o = out + (size_t)row * D_MODEL;
#pragma unroll
  for (int i = 0; i < 16; ++i) {
    int c = lane + 32 * i;
    o[c] = (_Float16)((v[i] - mean) * rstd * g[c] + bn[c]);
  }
}

// ---------------------------------------------------------------------------
// Edge bias precompute, TRANSPOSED output:
//   biasT[b,h,j,i] = edge[b,i,j,:] . we[h,:] + we_b[h]
// One thread per (b,i,j); edge read once (268 MB, coalesced 64B/thread).
// The [j][i] output order makes the attention-side reads contiguous 16B
// vectors per lane (C-fragment rows are contiguous in i). Scattered 2B
// stores here aggregate in the 192MB L2 before writeback.
// ---------------------------------------------------------------------------
__global__ __launch_bounds__(256) void edge_bias_kernel(
    const float* __restrict__ e, const float* __restrict__ we,
    const float* __restrict__ web, _Float16* __restrict__ biasT) {
  size_t idx = (size_t)blockIdx.x * blockDim.x + threadIdx.x;  // [0, B*N*N)
  const float4* ep = (const float4*)(e) + idx * 4;
  float4 e0 = ep[0], e1 = ep[1], e2 = ep[2], e3 = ep[3];
  float ev[16] = {e0.x, e0.y, e0.z, e0.w, e1.x, e1.y, e1.z, e1.w,
                  e2.x, e2.y, e2.z, e2.w, e3.x, e3.y, e3.z, e3.w};
  size_t b = idx >> 20;               // N*N == 1<<20
  size_t i = (idx >> 10) & 1023;
  size_t j = idx & 1023;
#pragma unroll
  for (int h = 0; h < 8; ++h) {
    float a = web[h];
#pragma unroll
    for (int k = 0; k < 16; ++k) a += we[h * 16 + k] * ev[k];
    biasT[(((size_t)b * 8 + h) << 20) + j * NTOK + i] = (_Float16)a;
  }
}

// ---------------------------------------------------------------------------
// Generic WMMA GEMM:  D[M,Nout] = A[M,K](f16) x W[Nout,K](f16)^T  (+ epilogue)
// One wave computes a 32x64 tile (2 A-frags x 4 B-frags -> 8 WMMAs/k-step).
// MODE selects epilogue:
//   0: f16 out scattered to [B,H,token,dk]          (Q, K)
//   1: f16 out scattered transposed to [B,H,dk,tok] (V^T, packed 16B stores)
//   2: f32 out = acc + bias[col] + resid[row,col]   (WO, FFN2)
//   3: f16 out = gelu_exact(acc + bias[col])        (FFN1)
// ---------------------------------------------------------------------------
template <int MODE>
__global__ __launch_bounds__(256) void gemm_kernel(
    const _Float16* __restrict__ A, const _Float16* __restrict__ W,
    const float* __restrict__ bias, const float* __restrict__ resid,
    void* __restrict__ outp, int M, int Nout, int K) {
  const int lane = threadIdx.x & 31;
  const int wid = blockIdx.x * (blockDim.x >> 5) + (threadIdx.x >> 5);
  const int ntx = Nout >> 6;
  const int m0 = (wid / ntx) << 5;   // 32-row wave tile
  const int n0 = (wid % ntx) << 6;   // 64-col wave tile
  if (m0 >= M) return;

  v8f acc[2][4] = {};
  for (int k0 = 0; k0 < K; k0 += 32) {
    v16h a0 = load_a16(A, K, m0, k0, lane);
    v16h a1 = load_a16(A, K, m0 + 16, k0, lane);
#pragma unroll
    for (int t = 0; t < 4; ++t) {
      v16h b = load_b16(W, K, n0 + 16 * t, k0, lane);
      acc[0][t] = wmma_f16(a0, b, acc[0][t]);
      acc[1][t] = wmma_f16(a1, b, acc[1][t]);
    }
  }

  const int hi = (lane >> 4) & 1;
  const int nc = lane & 15;

  if (MODE == 0) {  // [B,H,token,dk] f16
    _Float16* o = (_Float16*)outp;
#pragma unroll
    for (int mi = 0; mi < 2; ++mi)
#pragma unroll
      for (int t = 0; t < 4; ++t) {
        int col = n0 + 16 * t + nc, hh = col >> 6, dd = col & 63;
#pragma unroll
        for (int r = 0; r < 8; ++r) {
          int row = m0 + 16 * mi + r + 8 * hi, bb = row >> 10, tok = row & 1023;
          o[(((size_t)bb * 8 + hh) * NTOK + tok) * DK + dd] = (_Float16)acc[mi][t][r];
        }
      }
  } else if (MODE == 1) {  // [B,H,dk,token] f16, packed 8-token stores
    _Float16* o = (_Float16*)outp;
    int bb = m0 >> 10;
#pragma unroll
    for (int mi = 0; mi < 2; ++mi) {
      int tl = (m0 & 1023) + 16 * mi + 8 * hi;
#pragma unroll
      for (int t = 0; t < 4; ++t) {
        int col = n0 + 16 * t + nc, hh = col >> 6, dd = col & 63;
        v8h pk;
#pragma unroll
        for (int r = 0; r < 8; ++r) pk[r] = (_Float16)acc[mi][t][r];
        *(v8h*)(o + (((size_t)bb * 8 + hh) * DK + dd) * NTOK + tl) = pk;
      }
    }
  } else if (MODE == 2) {  // f32 + bias + residual
    float* o = (float*)outp;
#pragma unroll
    for (int mi = 0; mi < 2; ++mi)
#pragma unroll
      for (int t = 0; t < 4; ++t) {
        int col = n0 + 16 * t + nc;
        float bv = bias ? bias[col] : 0.f;
#pragma unroll
        for (int r = 0; r < 8; ++r) {
          int row = m0 + 16 * mi + r + 8 * hi;
          float v = acc[mi][t][r] + bv;
          if (resid) v += resid[(size_t)row * Nout + col];
          o[(size_t)row * Nout + col] = v;
        }
      }
  } else {  // MODE 3: f16 = exact GELU(acc + bias)
    _Float16* o = (_Float16*)outp;
#pragma unroll
    for (int mi = 0; mi < 2; ++mi)
#pragma unroll
      for (int t = 0; t < 4; ++t) {
        int col = n0 + 16 * t + nc;
        float bv = bias[col];
#pragma unroll
        for (int r = 0; r < 8; ++r) {
          int row = m0 + 16 * mi + r + 8 * hi;
          float v = acc[mi][t][r] + bv;
          v = 0.5f * v * (1.0f + erff(v * 0.70710678118654752f));
          o[(size_t)row * Nout + col] = (_Float16)v;
        }
      }
  }
}

// ---------------------------------------------------------------------------
// Flash attention: one wave owns 16 query rows of one (b,h).
// S tile 16x32 via WMMA (Q x K^T), + transposed-bias (16B vector loads),
// mask, online softmax, then P x V with P relaid C->A layout through LDS.
// ---------------------------------------------------------------------------
__global__ __launch_bounds__(256) void attn_kernel(
    const _Float16* __restrict__ Qb, const _Float16* __restrict__ Kb,
    const _Float16* __restrict__ Vtb, const _Float16* __restrict__ biasT,
    const unsigned char* __restrict__ mask, _Float16* __restrict__ outb) {
  __shared__ __align__(16) _Float16 plds[8][16][40];  // 40-half rows: 80B = 5*16B
  const int lane = threadIdx.x & 31;
  const int w = threadIdx.x >> 5;
  const int wid = blockIdx.x * 8 + w;
  const int i0 = (wid & 63) << 4;
  const int bh = wid >> 6;
  const int b = bh >> 3, h = bh & 7;
  const int hi = (lane >> 4) & 1, nc = lane & 15;

  const _Float16* Qp = Qb + (size_t)bh * NTOK * DK;
  const _Float16* Kp = Kb + (size_t)bh * NTOK * DK;
  const _Float16* Vp = Vtb + (size_t)bh * DK * NTOK;
  const _Float16* Bp = biasT + ((size_t)bh << 20);
  const unsigned char* mp = mask + (size_t)b * NTOK;
  const float NEG_INF = -__builtin_inff();

  v16h qa0 = load_a16(Qp, DK, i0, 0, lane);
  v16h qa1 = load_a16(Qp, DK, i0, 32, lane);

  v8f o[4] = {};
  float mrow[8], lrow[8];
#pragma unroll
  for (int r = 0; r < 8; ++r) { mrow[r] = NEG_INF; lrow[r] = 0.f; }

  for (int j0 = 0; j0 < NTOK; j0 += 32) {
    const int jA = j0 + nc, jB = j0 + 16 + nc;
    // transposed bias: 8 contiguous rows (i) per lane -> single 16B loads
    v8h bja = *(const v8h*)(Bp + (size_t)jA * NTOK + i0 + 8 * hi);
    v8h bjb = *(const v8h*)(Bp + (size_t)jB * NTOK + i0 + 8 * hi);
    v8f s[2] = {};
    {
      v16h kb0 = load_b16(Kp, DK, j0, 0, lane);
      v16h kb1 = load_b16(Kp, DK, j0, 32, lane);
      s[0] = wmma_f16(qa0, kb0, s[0]);
      s[0] = wmma_f16(qa1, kb1, s[0]);
      v16h kb2 = load_b16(Kp, DK, j0 + 16, 0, lane);
      v16h kb3 = load_b16(Kp, DK, j0 + 16, 32, lane);
      s[1] = wmma_f16(qa0, kb2, s[1]);
      s[1] = wmma_f16(qa1, kb3, s[1]);
    }
    const bool ma = mp[jA] != 0, mb = mp[jB] != 0;
#pragma unroll
    for (int r = 0; r < 8; ++r) {
      float sa = s[0][r] * 0.125f + (float)bja[r];
      float sb = s[1][r] * 0.125f + (float)bjb[r];
      s[0][r] = ma ? sa : NEG_INF;
      s[1][r] = mb ? sb : NEG_INF;
    }
    float p0[8], p1[8];
#pragma unroll
    for (int r = 0; r < 8; ++r) {
      float rm = fmaxf(s[0][r], s[1][r]);
#pragma unroll
      for (int msk = 1; msk < 16; msk <<= 1) rm = fmaxf(rm, __shfl_xor(rm, msk, 32));
      float mn = fmaxf(mrow[r], rm);
      bool dead = (mn == NEG_INF);  // fully masked so far
      float alpha = dead ? 1.f : __expf(mrow[r] - mn);
      float a0 = dead ? 0.f : __expf(s[0][r] - mn);
      float a1 = dead ? 0.f : __expf(s[1][r] - mn);
      mrow[r] = mn;
      float ps = a0 + a1;
#pragma unroll
      for (int msk = 1; msk < 16; msk <<= 1) ps += __shfl_xor(ps, msk, 32);
      lrow[r] = lrow[r] * alpha + ps;
      p0[r] = a0;
      p1[r] = a1;
      o[0][r] *= alpha; o[1][r] *= alpha; o[2][r] *= alpha; o[3][r] *= alpha;
    }
    // C-layout -> A-layout relayout of P through LDS
#pragma unroll
    for (int r = 0; r < 8; ++r) {
      plds[w][r + 8 * hi][nc] = (_Float16)p0[r];
      plds[w][r + 8 * hi][nc + 16] = (_Float16)p1[r];
    }
    __syncthreads();
    v16h pa = load_a16(&plds[w][0][0], 40, 0, 0, lane);
    __syncthreads();
#pragma unroll
    for (int t = 0; t < 4; ++t) {
      v16h vb = load_b16(Vp, NTOK, 16 * t, j0, lane);  // V^T rows contiguous
      o[t] = wmma_f16(pa, vb, o[t]);
    }
  }

#pragma unroll
  for (int r = 0; r < 8; ++r) {
    float inv = (lrow[r] > 0.f) ? 1.f / lrow[r] : 0.f;  // nan_to_num for dead rows
    o[0][r] *= inv; o[1][r] *= inv; o[2][r] *= inv; o[3][r] *= inv;
  }
#pragma unroll
  for (int t = 0; t < 4; ++t)
#pragma unroll
    for (int r = 0; r < 8; ++r) {
      int token = i0 + r + 8 * hi;
      int col = h * DK + 16 * t + nc;
      outb[((size_t)b * NTOK + token) * D_MODEL + col] = (_Float16)o[t][r];
    }
}

// ---------------------------------------------------------------------------
// Host orchestration
// ---------------------------------------------------------------------------
extern "C" void kernel_launch(void* const* d_in, const int* in_sizes, int n_in,
                              void* d_out, int out_size, void* d_ws,
                              size_t ws_size, hipStream_t stream) {
  (void)in_sizes; (void)n_in; (void)out_size; (void)ws_size;
  const float* x = (const float*)d_in[0];
  const float* edge = (const float*)d_in[1];
  const unsigned char* mask = (const unsigned char*)d_in[2];
  const float* wq = (const float*)d_in[3];
  const float* wk = (const float*)d_in[4];
  const float* wv = (const float*)d_in[5];
  const float* wo = (const float*)d_in[6];
  const float* wo_b = (const float*)d_in[7];
  const float* we = (const float*)d_in[8];
  const float* we_b = (const float*)d_in[9];
  const float* g1 = (const float*)d_in[10];
  const float* b1n = (const float*)d_in[11];
  const float* g2 = (const float*)d_in[12];
  const float* b2n = (const float*)d_in[13];
  const float* w1 = (const float*)d_in[14];
  const float* b1 = (const float*)d_in[15];
  const float* w2 = (const float*)d_in[16];
  const float* b2 = (const float*)d_in[17];
  float* out = (float*)d_out;

  char* ws = (char*)d_ws;
  size_t off = 0;
  auto alloc = [&](size_t bytes) {
    void* p = ws + off;
    off += (bytes + 255) & ~(size_t)255;
    return p;
  };
  _Float16* xn_h  = (_Float16*)alloc((size_t)MROWS * D_MODEL * 2);
  _Float16* wq_h  = (_Float16*)alloc((size_t)D_MODEL * D_MODEL * 2);
  _Float16* wk_h  = (_Float16*)alloc((size_t)D_MODEL * D_MODEL * 2);
  _Float16* wv_h  = (_Float16*)alloc((size_t)D_MODEL * D_MODEL * 2);
  _Float16* wo_h  = (_Float16*)alloc((size_t)D_MODEL * D_MODEL * 2);
  _Float16* w1_h  = (_Float16*)alloc((size_t)DFF * D_MODEL * 2);
  _Float16* w2_h  = (_Float16*)alloc((size_t)D_MODEL * DFF * 2);
  _Float16* Qb    = (_Float16*)alloc((size_t)MROWS * D_MODEL * 2);
  _Float16* Kbuf  = (_Float16*)alloc((size_t)MROWS * D_MODEL * 2);
  _Float16* Vtb   = (_Float16*)alloc((size_t)MROWS * D_MODEL * 2);
  _Float16* attn_h= (_Float16*)alloc((size_t)MROWS * D_MODEL * 2);
  float*    x1    = (float*)   alloc((size_t)MROWS * D_MODEL * 4);
  _Float16* xn2_h = (_Float16*)alloc((size_t)MROWS * D_MODEL * 2);
  _Float16* h_h   = (_Float16*)alloc((size_t)MROWS * DFF * 2);
  _Float16* biasT = (_Float16*)alloc((size_t)BATCH * NHEADS * NTOK * NTOK * 2);

  auto cvt = [&](const float* src, _Float16* dst, int n) {
    f32_to_f16_kernel<<<(n + 255) / 256, 256, 0, stream>>>(src, dst, n);
  };
  cvt(wq, wq_h, D_MODEL * D_MODEL);
  cvt(wk, wk_h, D_MODEL * D_MODEL);
  cvt(wv, wv_h, D_MODEL * D_MODEL);
  cvt(wo, wo_h, D_MODEL * D_MODEL);
  cvt(w1, w1_h, DFF * D_MODEL);
  cvt(w2, w2_h, D_MODEL * DFF);

  // LN1 -> xn (f16)
  ln_kernel<<<MROWS / 8, 256, 0, stream>>>(x, g1, b1n, xn_h, MROWS);

  // Q, K ([B,H,tok,dk]) and V^T ([B,H,dk,tok]); 32x64 wave tiles
  const int blk512 = (MROWS / 32) * (D_MODEL / 64) / 8;  // 128 blocks
  gemm_kernel<0><<<blk512, 256, 0, stream>>>(xn_h, wq_h, nullptr, nullptr, Qb,
                                             MROWS, D_MODEL, D_MODEL);
  gemm_kernel<0><<<blk512, 256, 0, stream>>>(xn_h, wk_h, nullptr, nullptr, Kbuf,
                                             MROWS, D_MODEL, D_MODEL);
  gemm_kernel<1><<<blk512, 256, 0, stream>>>(xn_h, wv_h, nullptr, nullptr, Vtb,
                                             MROWS, D_MODEL, D_MODEL);

  // Per-head additive edge bias (single pass over edge_emb), transposed layout
  edge_bias_kernel<<<(BATCH * NTOK * NTOK) / 256, 256, 0, stream>>>(edge, we,
                                                                    we_b, biasT);

  // Flash attention: B*H*(N/16) = 2048 waves
  attn_kernel<<<256, 256, 0, stream>>>(Qb, Kbuf, Vtb, biasT, mask, attn_h);

  // x1 = x + attn @ wo^T + wo_b   (f32)
  gemm_kernel<2><<<blk512, 256, 0, stream>>>(attn_h, wo_h, wo_b, x, x1, MROWS,
                                             D_MODEL, D_MODEL);

  // LN2 -> xn2 (f16)
  ln_kernel<<<MROWS / 8, 256, 0, stream>>>(x1, g2, b2n, xn2_h, MROWS);

  // h = gelu(xn2 @ w1^T + b1)   (f16)
  const int blkFF = (MROWS / 32) * (DFF / 64) / 8;  // 512 blocks
  gemm_kernel<3><<<blkFF, 256, 0, stream>>>(xn2_h, w1_h, b1, nullptr, h_h,
                                            MROWS, DFF, D_MODEL);

  // out = x1 + h @ w2^T + b2    (f32)
  gemm_kernel<2><<<blk512, 256, 0, stream>>>(h_h, w2_h, b2, x1, out, MROWS,
                                             D_MODEL, DFF);
}
